// Transformer_83245056131968
// MI455X (gfx1250) — matrix operand order repaired and checked by reading, not compiled, not run on hardware
//
#include <hip/hip_runtime.h>
#include <hip/hip_bf16.h>
#include <math.h>

typedef unsigned short u16;
typedef unsigned int   u32;
typedef __attribute__((ext_vector_type(16))) __bf16 bf16x16;
typedef __attribute__((ext_vector_type(8)))  float  f32x8;
typedef int i32x4 __attribute__((vector_size(16)));
typedef __attribute__((address_space(1))) i32x4 as1_i32x4;
typedef __attribute__((address_space(3))) i32x4 as3_i32x4;

union FragB16 { bf16x16 v; u32 u[8]; u16 s[16]; };

constexpr int B_ = 8, N_ = 1024, D_ = 256, H_ = 8, L_ = 6;
constexpr int T_ = N_ + 2;       // 1026
constexpr int M_ = B_ * T_;      // 8208
constexpr int MP_ = 8320;        // 65 * 128, padded rows for guard-free tile loads
constexpr float INV_SQRT_D = 0.0625f;

#if __has_builtin(__builtin_amdgcn_global_load_async_to_lds_b128) && \
    __has_builtin(__builtin_amdgcn_s_wait_asynccnt)
#define USE_ASYNC 1
#define ASYNC_WAIT(n) __builtin_amdgcn_s_wait_asynccnt(n)
#else
#define USE_ASYNC 0
#define ASYNC_WAIT(n)
#endif

__device__ __forceinline__ u16 f2bf(float f) {
  u32 u = __float_as_uint(f);
  u32 r = u + 0x7fffu + ((u >> 16) & 1u);  // round-to-nearest-even
  return (u16)(r >> 16);
}
// 16-bit A-matrix 16x32 per-lane K mapping (ISA 7.12.2)
__device__ __forceinline__ int a_k0(int j, int h) {
  return ((j < 4) ? 2 * j : 2 * j + 8) + 8 * h;
}
__device__ __forceinline__ float gelu_exact(float v) {
  return 0.5f * v * (1.0f + erff(v * 0.70710678118654752440f));
}

// ---------------- weight convert + transpose: [K,N] f32 -> [N,K] bf16 ----------------
__global__ void k_w2bf_t(const float* __restrict__ in, u16* __restrict__ out,
                         int K, int N) {
  int i = blockIdx.x * blockDim.x + threadIdx.x;
  if (i >= K * N) return;
  int kk = i / N, n = i - kk * N;
  out[(size_t)n * K + kk] = f2bf(in[i]);
}

__global__ void k_build_h(const float* __restrict__ x, const float* __restrict__ cls,
                          const float* __restrict__ dict, float* __restrict__ h) {
  int i = blockIdx.x * blockDim.x + threadIdx.x;
  if (i >= B_ * T_ * D_) return;
  int d = i % D_;
  int t = (i / D_) % T_;
  int b = i / (D_ * T_);
  float v;
  if (t < N_)       v = x[((size_t)b * N_ + t) * D_ + d];
  else if (t == N_) v = cls[d];
  else              v = dict[d];
  h[i] = v;
}

// one wave32 per row of 256
__global__ __launch_bounds__(256)
void k_ln(const float* __restrict__ h, const float* __restrict__ g,
          const float* __restrict__ bta, u16* __restrict__ out) {
  int row  = blockIdx.x * 8 + (threadIdx.x >> 5);
  int lane = threadIdx.x & 31;
  if (row >= M_) return;
  const float* hr = h + (size_t)row * D_;
  float xv[8];
  float s = 0.f, ss = 0.f;
  for (int i = 0; i < 8; ++i) {
    float v = hr[lane + i * 32];
    xv[i] = v; s += v; ss += v * v;
  }
  for (int m = 16; m >= 1; m >>= 1) {
    s  += __shfl_xor(s,  m, 32);
    ss += __shfl_xor(ss, m, 32);
  }
  float mean = s * (1.0f / D_);
  float var  = ss * (1.0f / D_) - mean * mean;
  float rstd = rsqrtf(var + 1e-5f);
  u16* outr = out + (size_t)row * D_;
  for (int i = 0; i < 8; ++i) {
    int c = lane + i * 32;
    outr[c] = f2bf((xv[i] - mean) * rstd * g[c] + bta[c]);
  }
}

__global__ void k_scatter(const float* __restrict__ h, float* __restrict__ out) {
  int i = blockIdx.x * blockDim.x + threadIdx.x;
  if (i >= B_ * T_ * D_) return;
  int d = i % D_;
  int t = (i / D_) % T_;
  int b = i / (D_ * T_);
  float v = h[i];
  size_t base1 = (size_t)B_ * N_ * D_;
  if (t < N_)       out[((size_t)b * N_ + t) * D_ + d] = v;
  else if (t == N_) out[base1 + (size_t)b * D_ + d] = v;
  else              out[base1 + (size_t)B_ * D_ + (size_t)b * D_ + d] = v;
}

// ---------------- bf16 WMMA GEMM: C[M,N] = A[M,K] @ W^T + bias ----------------
// A: [MP,K] bf16 row-major (padded rows, no load guards)
// Bw: [N,K] bf16 (pre-transposed weights)
// mode 0: plain, 1: GELU, 2: QKV split into q/k/v [B,H,T,256] (q,k scaled 1/sqrt(D))
__global__ __launch_bounds__(256)
void k_gemm(const u16* __restrict__ A, const u16* __restrict__ Bw,
            const float* __restrict__ bias, float* __restrict__ Cf,
            u16* __restrict__ Cb, int M, int K, int N, int mode,
            u16* __restrict__ qo, u16* __restrict__ ko, u16* __restrict__ vo) {
  __shared__ __attribute__((aligned(16))) u16 Al[2][128 * 32];
  __shared__ __attribute__((aligned(16))) u16 Bl[2][128 * 32];
  int tid  = threadIdx.x;
  int lane = tid & 31, wid = tid >> 5;
  int wr = wid & 3, wc = wid >> 2;
  int lrow = lane & 15, lh = lane >> 4;
  int m0 = blockIdx.y * 128, n0 = blockIdx.x * 128;

  f32x8 acc[2][4];
  for (int mi = 0; mi < 2; ++mi)
    for (int ni = 0; ni < 4; ++ni)
      for (int j = 0; j < 8; ++j) acc[mi][ni][j] = 0.f;

  // stage one 128x32 panel of A and of W^T into LDS buffer `buf`
  auto stage = [&](int kb, int buf) {
    for (int i = 0; i < 2; ++i) {
      int idx = tid + i * 256;
      int ar = idx >> 2, kk = (idx & 3) * 8;
      const u16* ga = A  + (size_t)(m0 + ar) * K + kb + kk;
      const u16* gb = Bw + (size_t)(n0 + ar) * K + kb + kk;
      u16* la = &Al[buf][ar * 32 + kk];
      u16* lb = &Bl[buf][ar * 32 + kk];
#if USE_ASYNC
      __builtin_amdgcn_global_load_async_to_lds_b128(
          (as1_i32x4*)(void*)ga, (as3_i32x4*)la, 0, 0);
      __builtin_amdgcn_global_load_async_to_lds_b128(
          (as1_i32x4*)(void*)gb, (as3_i32x4*)lb, 0, 0);
#else
      *(uint4*)la = *(const uint4*)ga;
      *(uint4*)lb = *(const uint4*)gb;
#endif
    }
  };
  auto compute = [&](int buf) {
    FragB16 af[2];
    for (int mi = 0; mi < 2; ++mi) {
      int r = wr * 32 + mi * 16 + lrow;
      for (int j = 0; j < 8; ++j)
        af[mi].u[j] = *(const u32*)&Al[buf][r * 32 + a_k0(j, lh)];
    }
    for (int ni = 0; ni < 4; ++ni) {
      FragB16 bf;
      int n = wc * 64 + ni * 16 + lrow;
      for (int j = 0; j < 8; ++j)
        bf.u[j] = *(const u32*)&Bl[buf][n * 32 + a_k0(j, lh)];
      for (int mi = 0; mi < 2; ++mi)
        acc[mi][ni] = __builtin_amdgcn_wmma_f32_16x16x32_bf16(
            false, af[mi].v, false, bf.v, (short)0, acc[mi][ni], false, false);
    }
  };

  const int nT = K >> 5;
  stage(0, 0);
  for (int t = 0; t < nT - 1; ++t) {
    stage((t + 1) << 5, (t + 1) & 1);
    ASYNC_WAIT(4);          // previous tile's 4 async copies complete (in-order)
    __syncthreads();
    compute(t & 1);
    __syncthreads();        // all waves done reading before buffer reuse
  }
  ASYNC_WAIT(0);
  __syncthreads();
  compute((nT - 1) & 1);

  // epilogue
  for (int mi = 0; mi < 2; ++mi) {
    for (int ni = 0; ni < 4; ++ni) {
      int col = n0 + wc * 64 + ni * 16 + lrow;
      float bv = bias ? bias[col] : 0.f;
      for (int j = 0; j < 8; ++j) {
        int row = m0 + wr * 32 + mi * 16 + j + 8 * lh;
        if (row >= M) continue;
        float v = acc[mi][ni][j] + bv;
        if (mode == 1) v = gelu_exact(v);
        if (mode == 2) {
          int b = row / T_, t = row - b * T_;
          int hh = col / (D_ * 3);
          int rem = col - hh * (D_ * 3);
          int f = rem / 3, comp = rem - f * 3;
          size_t dst = (((size_t)(b * H_ + hh)) * T_ + t) * D_ + f;
          if (comp == 0)      qo[dst] = f2bf(v * INV_SQRT_D);
          else if (comp == 1) ko[dst] = f2bf(v * INV_SQRT_D);
          else                vo[dst] = f2bf(v);
        } else {
          size_t o = (size_t)row * N + col;
          if (Cf) Cf[o] = v;
          if (Cb) Cb[o] = f2bf(v);
        }
      }
    }
  }
}

// ---------------- flash attention, head dim 256 ----------------
__global__ __launch_bounds__(256)
void k_attn(const u16* __restrict__ q, const u16* __restrict__ k,
            const u16* __restrict__ v, u16* __restrict__ o) {
  __shared__ __attribute__((aligned(16))) float Sm[16 * 32];
  __shared__ __attribute__((aligned(16))) u16   Pm[16 * 32];
  __shared__ float rm[16], rl[16], rsc[16];
  int tid = threadIdx.x, lane = tid & 31, wid = tid >> 5;
  int lrow = lane & 15, lh = lane >> 4;
  int bh = blockIdx.x;
  int q0 = blockIdx.y * 16;
  const u16* qb = q + (size_t)bh * T_ * D_;
  const u16* kb = k + (size_t)bh * T_ * D_;
  const u16* vb = v + (size_t)bh * T_ * D_;
  int wdim = wid * 32;

  FragB16 qf;
  {
    int qr = q0 + lrow; if (qr > T_ - 1) qr = T_ - 1;
    const u16* qp = qb + (size_t)qr * D_ + wdim;
    for (int j = 0; j < 8; ++j) qf.u[j] = *(const u32*)&qp[a_k0(j, lh)];
  }
  f32x8 acc[2];
  for (int ni = 0; ni < 2; ++ni)
    for (int j = 0; j < 8; ++j) acc[ni][j] = 0.f;
  if (tid < 16) { rm[tid] = -3.0e38f; rl[tid] = 0.f; }
  __syncthreads();

  const int nkt = (T_ + 31) / 32;  // 33
  for (int kt = 0; kt < nkt; ++kt) {
    Sm[tid] = 0.f; Sm[tid + 256] = 0.f;
    __syncthreads();
    for (int kg = 0; kg < 2; ++kg) {
      FragB16 kf;
      int key = kt * 32 + kg * 16 + lrow; if (key > T_ - 1) key = T_ - 1;
      const u16* kp = kb + (size_t)key * D_ + wdim;
      for (int j = 0; j < 8; ++j) kf.u[j] = *(const u32*)&kp[a_k0(j, lh)];
      f32x8 sp;
      for (int j = 0; j < 8; ++j) sp[j] = 0.f;
      sp = __builtin_amdgcn_wmma_f32_16x16x32_bf16(
          false, qf.v, false, kf.v, (short)0, sp, false, false);
      for (int j = 0; j < 8; ++j)
        atomicAdd(&Sm[(j + 8 * lh) * 32 + kg * 16 + lrow], sp[j]);
    }
    __syncthreads();
    if (tid < 16) {
      int r = tid;
      float mold = rm[r], mnew = mold;
      for (int c = 0; c < 32; ++c) {
        float s = (kt * 32 + c < T_) ? Sm[r * 32 + c] : -3.0e38f;
        if (s > mnew) mnew = s;
      }
      float sc = __expf(mold - mnew);
      float l = rl[r] * sc;
      for (int c = 0; c < 32; ++c) {
        float p = (kt * 32 + c < T_) ? __expf(Sm[r * 32 + c] - mnew) : 0.f;
        l += p;
        Pm[r * 32 + c] = f2bf(p);
      }
      rm[r] = mnew; rl[r] = l; rsc[r] = sc;
    }
    __syncthreads();
    FragB16 pf;
    for (int j = 0; j < 8; ++j) pf.u[j] = *(const u32*)&Pm[lrow * 32 + a_k0(j, lh)];
    for (int ni = 0; ni < 2; ++ni) {
      for (int j = 0; j < 8; ++j) acc[ni][j] *= rsc[j + 8 * lh];
      FragB16 vf;
      int dim = wdim + ni * 16 + lrow;
      for (int j = 0; j < 8; ++j) {
        int k0 = kt * 32 + a_k0(j, lh);
        int k1 = k0 + 1;
        if (k0 > T_ - 1) k0 = T_ - 1;
        if (k1 > T_ - 1) k1 = T_ - 1;
        u32 lo = vb[(size_t)k0 * D_ + dim];
        u32 hi = vb[(size_t)k1 * D_ + dim];
        vf.u[j] = lo | (hi << 16);
      }
      acc[ni] = __builtin_amdgcn_wmma_f32_16x16x32_bf16(
          false, pf.v, false, vf.v, (short)0, acc[ni], false, false);
    }
    __syncthreads();
  }
  int b = bh >> 3, hh = bh & 7;
  for (int ni = 0; ni < 2; ++ni) {
    int dim = wdim + ni * 16 + lrow;
    for (int j = 0; j < 8; ++j) {
      int t = q0 + j + 8 * lh;
      if (t >= T_) continue;
      float inv = 1.0f / rl[j + 8 * lh];
      o[((size_t)(b * T_ + t)) * (H_ * D_) + hh * D_ + dim] = f2bf(acc[ni][j] * inv);
    }
  }
}

// ---------------- host launch ----------------
extern "C" void kernel_launch(void* const* d_in, const int* in_sizes, int n_in,
                              void* d_out, int out_size, void* d_ws, size_t ws_size,
                              hipStream_t stream) {
  (void)in_sizes; (void)n_in; (void)out_size; (void)ws_size;
  const float* x      = (const float*)d_in[0];
  const float* cls    = (const float*)d_in[1];
  const float* dict   = (const float*)d_in[2];
  const float* ln1_g  = (const float*)d_in[3];
  const float* ln1_b  = (const float*)d_in[4];
  const float* qkv_w  = (const float*)d_in[5];
  const float* qkv_b  = (const float*)d_in[6];
  const float* proj_w = (const float*)d_in[7];
  const float* proj_b = (const float*)d_in[8];
  const float* ln2_g  = (const float*)d_in[9];
  const float* ln2_b  = (const float*)d_in[10];
  const float* mlp_w1 = (const float*)d_in[11];
  const float* mlp_b1 = (const float*)d_in[12];
  const float* mlp_w2 = (const float*)d_in[13];
  const float* mlp_b2 = (const float*)d_in[14];

  char* wp = (char*)d_ws;
  size_t off = 0;
  auto alloc = [&](size_t bytes) -> void* {
    off = (off + 255) & ~(size_t)255;
    void* p = wp + off;
    off += bytes;
    return p;
  };
  const size_t szQkv = (size_t)D_ * H_ * D_ * 3;  // per-layer elems
  const size_t szPr  = (size_t)H_ * D_ * D_;
  const size_t szW1  = (size_t)D_ * 4 * D_;
  const size_t szW2  = (size_t)4 * D_ * D_;
  u16*   Wqkv = (u16*)  alloc(L_ * szQkv * 2);   // [N=6144][K=256] per layer
  u16*   Wpr  = (u16*)  alloc(L_ * szPr * 2);    // [N=256][K=2048]
  u16*   W1   = (u16*)  alloc(L_ * szW1 * 2);    // [N=1024][K=256]
  u16*   W2   = (u16*)  alloc(L_ * szW2 * 2);    // [N=256][K=1024]
  float* h    = (float*)alloc((size_t)MP_ * D_ * 4);
  float* h2   = (float*)alloc((size_t)MP_ * D_ * 4);
  u16*   hn   = (u16*)  alloc((size_t)MP_ * D_ * 2);
  u16*   qbuf = (u16*)  alloc((size_t)B_ * H_ * T_ * D_ * 2);
  u16*   kbuf = (u16*)  alloc((size_t)B_ * H_ * T_ * D_ * 2);
  u16*   vbuf = (u16*)  alloc((size_t)B_ * H_ * T_ * D_ * 2);
  u16*   obuf = (u16*)  alloc((size_t)MP_ * H_ * D_ * 2);
  u16*   m1   = (u16*)  alloc((size_t)MP_ * 4 * D_ * 2);

  // one-time weight convert + transpose to [N][K] bf16
  for (int l = 0; l < L_; ++l) {
    k_w2bf_t<<<(int)((szQkv + 255) / 256), 256, 0, stream>>>(
        qkv_w + l * szQkv, Wqkv + l * szQkv, D_, H_ * D_ * 3);
    k_w2bf_t<<<(int)((szPr + 255) / 256), 256, 0, stream>>>(
        proj_w + l * szPr, Wpr + l * szPr, H_ * D_, D_);
    k_w2bf_t<<<(int)((szW1 + 255) / 256), 256, 0, stream>>>(
        mlp_w1 + l * szW1, W1 + l * szW1, D_, 4 * D_);
    k_w2bf_t<<<(int)((szW2 + 255) / 256), 256, 0, stream>>>(
        mlp_w2 + l * szW2, W2 + l * szW2, 4 * D_, D_);
  }

  k_build_h<<<(B_ * T_ * D_ + 255) / 256, 256, 0, stream>>>(x, cls, dict, h);

  const int mTiles = MP_ / 128;  // 65
  for (int l = 0; l < L_; ++l) {
    k_ln<<<(M_ + 7) / 8, 256, 0, stream>>>(h, ln1_g + l * D_, ln1_b + l * D_, hn);
    k_gemm<<<dim3(6144 / 128, mTiles), 256, 0, stream>>>(
        hn, Wqkv + l * szQkv, qkv_b + (size_t)l * 6144,
        nullptr, nullptr, M_, D_, 6144, 2, qbuf, kbuf, vbuf);
    k_attn<<<dim3(B_ * H_, (T_ + 15) / 16), 256, 0, stream>>>(qbuf, kbuf, vbuf, obuf);
    k_gemm<<<dim3(256 / 128, mTiles), 256, 0, stream>>>(
        obuf, Wpr + l * szPr, proj_b + (size_t)l * 256,
        h2, nullptr, M_, 2048, 256, 0, nullptr, nullptr, nullptr);
    k_ln<<<(M_ + 7) / 8, 256, 0, stream>>>(h2, ln2_g + l * D_, ln2_b + l * D_, hn);
    k_gemm<<<dim3(1024 / 128, mTiles), 256, 0, stream>>>(
        hn, W1 + l * szW1, mlp_b1 + (size_t)l * 1024,
        nullptr, m1, M_, D_, 1024, 1, nullptr, nullptr, nullptr);
    k_gemm<<<dim3(256 / 128, mTiles), 256, 0, stream>>>(
        m1, W2 + l * szW2, mlp_b2 + (size_t)l * 256,
        h, nullptr, M_, 1024, 256, 1, nullptr, nullptr, nullptr);
  }
  k_scatter<<<(B_ * T_ * D_ + 255) / 256, 256, 0, stream>>>(h, (float*)d_out);
}
